// MultiHeadAttention_44478681317743
// MI455X (gfx1250) — matrix operand order repaired and checked by reading, not compiled, not run on hardware
//
#include <hip/hip_runtime.h>
#include <hip/hip_bf16.h>
#include <math.h>
#include <stdint.h>

// Problem constants (must match reference)
#define BB    2
#define SS    2048
#define HIDC  1024
#define NHC   16
#define HDC   64
#define CHUNKC 128

typedef __attribute__((ext_vector_type(16))) __bf16 v16bf;
typedef __attribute__((ext_vector_type(8)))  __bf16 v8bf;
typedef __attribute__((ext_vector_type(8)))  float  v8f;
typedef __attribute__((ext_vector_type(4)))  unsigned int v4u;
typedef __attribute__((ext_vector_type(8)))  int v8i;
typedef __attribute__((ext_vector_type(4)))  int v4i;

#ifndef __has_builtin
#define __has_builtin(x) 0
#endif
#if __has_builtin(__builtin_amdgcn_tensor_load_to_lds) && \
    __has_builtin(__builtin_amdgcn_s_wait_tensorcnt)
#define HAVE_TDM 1
#else
#define HAVE_TDM 0
#endif

// ---------------------------------------------------------------------------
// WMMA fragment helpers (CDNA5 16x16x32 bf16 layouts, CDNA5 ISA §7.12.2)
//  A (16x32, 16-bit): lane<16 -> row=lane, K={k..k+7, k+16..k+23}; lane>=16: +8
//  B (32x16, 16-bit): lane<16 -> col=lane, K={k..k+7, k+16..k+23}; lane>=16: +8
//  C/D (16x16 f32):   VGPR r -> M = r + 8*(lane/16), N = lane&15
// ---------------------------------------------------------------------------
static __device__ __forceinline__ v16bf load_a_frag(const __bf16* base, int stride,
                                                    int row0, int k, int lane) {
  const int r = lane & 15, hf = lane >> 4;
  const __bf16* p = base + (size_t)(row0 + r) * stride + k + (hf << 3);
  v8bf c0 = *(const v8bf*)p;
  v8bf c1 = *(const v8bf*)(p + 16);
  v16bf a;
#pragma unroll
  for (int i = 0; i < 8; ++i) { a[i] = c0[i]; a[i + 8] = c1[i]; }
  return a;
}

static __device__ __forceinline__ v16bf load_b_frag(const __bf16* base, int stride,
                                                    int col0, int k, int lane) {
  const int c = lane & 15, hf = lane >> 4;
  const __bf16* p = base + (size_t)(col0 + c) * stride + k + (hf << 3);
  v8bf c0 = *(const v8bf*)p;
  v8bf c1 = *(const v8bf*)(p + 16);
  v16bf b;
#pragma unroll
  for (int i = 0; i < 8; ++i) { b[i] = c0[i]; b[i + 8] = c1[i]; }
  return b;
}

static __device__ __forceinline__ v8f wmma_bf16(v16bf a, v16bf b, v8f c) {
  return __builtin_amdgcn_wmma_f32_16x16x32_bf16(false, a, false, b, (short)0, c,
                                                 false, false);
}

// ---------------------------------------------------------------------------
// Tensor Data Mover: DMA a 2D tile (bf16) from global into LDS.
// D# built per CDNA5 ISA:
//  group0: [1:0]=count=1, [63:32]=lds byte addr, [120:64]=global byte addr,
//          [127:126]=type=2
//  group1: [17:16]=data_size(1 => 2 bytes), [79:48]=tensor_dim0,
//          [111:80]=tensor_dim1, [127:112]=tile_dim0, [143:128]=tile_dim1,
//          [207:160]=tensor_dim0_stride (elements)
// Issued by one wave; completion via s_wait_tensorcnt + workgroup barrier.
// ---------------------------------------------------------------------------
static __device__ __forceinline__ void tdm_load_2d(uint32_t lds_off, const void* gptr,
                                                   uint32_t tensor_d0, uint32_t tensor_d1,
                                                   uint32_t tile_d0, uint32_t tile_d1,
                                                   uint64_t stride0_elems) {
#if HAVE_TDM
  const uint64_t ga = (uint64_t)(uintptr_t)gptr;
  v4u g0;
  g0[0] = 1u;                                            // count=1, user mode
  g0[1] = lds_off;                                       // LDS byte address
  g0[2] = (uint32_t)ga;                                  // global addr [95:64]
  g0[3] = (uint32_t)((ga >> 32) & 0x01FFFFFFu) | (2u << 30);  // addr hi | type=2
  v8i g1;
  g1[0] = (int)(1u << 16);                               // data_size = 2 bytes
  g1[1] = (int)((tensor_d0 & 0xFFFFu) << 16);            // tensor_dim0 lo
  g1[2] = (int)((tensor_d0 >> 16) | ((tensor_d1 & 0xFFFFu) << 16));
  g1[3] = (int)((tensor_d1 >> 16) | (tile_d0 << 16));    // tile_dim0
  g1[4] = (int)(tile_d1 & 0xFFFFu);                      // tile_dim1 (tile_dim2=0)
  g1[5] = (int)(uint32_t)stride0_elems;                  // tensor_dim0_stride lo
  g1[6] = (int)(uint32_t)((stride0_elems >> 32) & 0xFFFFu);
  g1[7] = 0;
  v4i z4 = (v4i)0;
#if __clang_major__ >= 23
  v8i z8 = (v8i)0;
  __builtin_amdgcn_tensor_load_to_lds(g0, g1, z4, z4, z8, 0);
#else
  __builtin_amdgcn_tensor_load_to_lds(g0, g1, z4, z4, 0);
#endif
#else
  (void)lds_off; (void)gptr; (void)tensor_d0; (void)tensor_d1;
  (void)tile_d0; (void)tile_d1; (void)stride0_elems;
#endif
}

static __device__ __forceinline__ void tdm_wait_sync() {
#if HAVE_TDM
  __builtin_amdgcn_s_wait_tensorcnt(0);
#endif
  __syncthreads();
}

// Fallback cooperative copy (used only if the TDM builtin is unavailable).
static __device__ __forceinline__ void stage_tile_fallback(__bf16* dst, const __bf16* src,
                                                           int rows, int rowlen,
                                                           int srcStride, int tid,
                                                           int nthreads) {
  const int chunks = rows * rowlen / 8;
  for (int i = tid; i < chunks; i += nthreads) {
    const int row = (i * 8) / rowlen, colc = (i * 8) % rowlen;
    *(v8bf*)(dst + row * rowlen + colc) =
        *(const v8bf*)(src + (size_t)row * srcStride + colc);
  }
}

// ---------------------------------------------------------------------------
// Kernel 1: f32 -> bf16 conversion
// ---------------------------------------------------------------------------
__global__ __launch_bounds__(256) void cvt_bf16_kernel(const float* __restrict__ src,
                                                       __bf16* __restrict__ dst, int n) {
  int i = blockIdx.x * blockDim.x + threadIdx.x;
  if (i < n) dst[i] = (__bf16)src[i];
}

// ---------------------------------------------------------------------------
// Kernel 2: QKV projection GEMM.  C(row,col) = sum_k hs(row,k) * W(col,k) + b(col)
// M = 4096, N = 1024, K = 1024. Block tile 128(M) x 64(N); wave tile 16 x 64.
// Weight panel (64 cols x 128 k) staged into LDS by TDM, double-buffered.
// mode 0/1: write (B,NH,S,HD) bf16 (Q/K).  mode 2: write V^T (B,NH,HD,S) bf16.
// ---------------------------------------------------------------------------
__global__ __launch_bounds__(256) void qkv_gemm_kernel(const __bf16* __restrict__ hsb,
                                                       const __bf16* __restrict__ wb,
                                                       const float* __restrict__ bias,
                                                       __bf16* __restrict__ dstQK,
                                                       __bf16* __restrict__ dstVT,
                                                       int mode) {
  __shared__ __bf16 __attribute__((aligned(64))) wstage[2][64 * 128];
  const int tid = threadIdx.x;
  const int lane = tid & 31, wave = tid >> 5;
  const int rowBase = blockIdx.x * 128 + wave * 16;
  const int colBase = blockIdx.y * 64;

  auto stage_w = [&](int buf, int kpanel) {
#if HAVE_TDM
    if (wave == 0)
      tdm_load_2d((uint32_t)(uintptr_t)&wstage[buf][0],
                  wb + (size_t)colBase * HIDC + kpanel * 128,
                  /*tensor_d0=*/128, /*tensor_d1=*/64,
                  /*tile_d0=*/128, /*tile_d1=*/64, /*stride0=*/HIDC);
#else
    stage_tile_fallback(&wstage[buf][0], wb + (size_t)colBase * HIDC + kpanel * 128,
                        64, 128, HIDC, tid, 256);
#endif
  };

  stage_w(0, 0);
  tdm_wait_sync();

  v8f acc[4] = {};
  for (int p = 0; p < HIDC / 128; ++p) {
    if (p + 1 < HIDC / 128) stage_w((p + 1) & 1, p + 1);
    const __bf16* wl = &wstage[p & 1][0];
#pragma unroll
    for (int kk = 0; kk < 128; kk += 32) {
      v16bf a = load_a_frag(hsb, HIDC, rowBase, p * 128 + kk, lane);
#pragma unroll
      for (int t = 0; t < 4; ++t) {
        v16bf b = load_b_frag(wl, 128, 16 * t, kk, lane);
        acc[t] = wmma_bf16(a, b, acc[t]);
      }
    }
    tdm_wait_sync();
  }

  const int lh = lane >> 4, n = lane & 15;
#pragma unroll
  for (int t = 0; t < 4; ++t) {
    const int col = colBase + 16 * t + n;
    const int hh = col >> 6, d = col & 63;
    const float bv = bias[col];
#pragma unroll
    for (int r = 0; r < 8; ++r) {
      const int row = rowBase + r + (lh << 3);
      const int bi = row >> 11, s = row & (SS - 1);
      const float v = acc[t][r] + bv;
      if (mode == 2)
        dstVT[(((size_t)(bi * NHC + hh)) * HDC + d) * SS + s] = (__bf16)v;
      else
        dstQK[(((size_t)(bi * NHC + hh)) * SS + s) * HDC + d] = (__bf16)v;
    }
  }
}

// ---------------------------------------------------------------------------
// Kernel 3: fused flash attention (online softmax), per wave: 16 q-rows.
// Grid (S/128, NH, B), 256 threads. K block and V^T block staged via TDM
// (double-buffered, prefetch overlapped with WMMA). Writes context + stats.
// ---------------------------------------------------------------------------
__global__ __launch_bounds__(256) void flash_attn_kernel(const __bf16* __restrict__ qb,
                                                         const __bf16* __restrict__ kbuf,
                                                         const __bf16* __restrict__ vtb,
                                                         const float* __restrict__ mask,
                                                         const float* __restrict__ hw,
                                                         float* __restrict__ ctx_out,
                                                         float* __restrict__ stats) {
  __shared__ __bf16 __attribute__((aligned(64))) kstage[2][128 * 64]; // keys x d
  __shared__ __bf16 __attribute__((aligned(64))) vstage[2][64 * 128]; // d x keys
  __shared__ __bf16 __attribute__((aligned(64))) pbuf[8 * 16 * 128];  // per-wave P
  const int tid = threadIdx.x;
  const int lane = tid & 31, wave = tid >> 5;
  const int qblk = blockIdx.x, hh = blockIdx.y, bi = blockIdx.z;
  const int bh = bi * NHC + hh;
  const int qBase = qblk * 128 + wave * 16;
  const int lh = lane >> 4, n = lane & 15;
  const __bf16* qp = qb + (size_t)bh * SS * HDC;
  const __bf16* kp = kbuf + (size_t)bh * SS * HDC;
  const __bf16* vp = vtb + (size_t)bh * HDC * SS;
  __bf16* pw = &pbuf[wave * (16 * 128)];

  auto stage_kv = [&](int buf, int kb) {
#if HAVE_TDM
    if (wave == 0)
      tdm_load_2d((uint32_t)(uintptr_t)&kstage[buf][0], kp + (size_t)kb * HDC,
                  /*tensor_d0=*/HDC, /*tensor_d1=*/128,
                  /*tile_d0=*/HDC, /*tile_d1=*/128, /*stride0=*/HDC);
    else if (wave == 1)
      tdm_load_2d((uint32_t)(uintptr_t)&vstage[buf][0], vp + kb,
                  /*tensor_d0=*/128, /*tensor_d1=*/HDC,
                  /*tile_d0=*/128, /*tile_d1=*/HDC, /*stride0=*/SS);
#else
    stage_tile_fallback(&kstage[buf][0], kp + (size_t)kb * HDC, 128, HDC, HDC, tid, 256);
    stage_tile_fallback(&vstage[buf][0], vp + kb, HDC, 128, SS, tid, 256);
#endif
  };

  v16bf qf[2];
#pragma unroll
  for (int j = 0; j < 2; ++j) qf[j] = load_a_frag(qp, HDC, qBase, 32 * j, lane);

  float m_run[8], l_run[8];
#pragma unroll
  for (int r = 0; r < 8; ++r) { m_run[r] = -1e30f; l_run[r] = 0.f; }
  v8f acc[4] = {};

  const float* mask_b = mask + (size_t)bi * SS * SS;

  stage_kv(0, 0);
  tdm_wait_sync();

  for (int it = 0; it < SS / 128; ++it) {
    const int kb0 = it * 128;
    if (it + 1 < SS / 128) stage_kv((it + 1) & 1, kb0 + 128);
    const __bf16* kbL = &kstage[it & 1][0];
    const __bf16* vbL = &vstage[it & 1][0];

    // ---- scores: 8 tiles of 16x16 over the 128 staged keys
    v8f st[8];
#pragma unroll
    for (int t = 0; t < 8; ++t) {
      v8f s = {};
#pragma unroll
      for (int j = 0; j < 2; ++j) {
        v16bf bf = load_b_frag(kbL, HDC, 16 * t, 32 * j, lane);
        s = wmma_bf16(qf[j], bf, s);
      }
      st[t] = s;
    }
    // ---- scale + mask + per-lane row max
    float mx[8];
#pragma unroll
    for (int r = 0; r < 8; ++r) {
      const int q = qBase + r + (lh << 3);
      const float* mrow = mask_b + (size_t)q * SS + kb0 + n;
      float rm = -1e30f;
#pragma unroll
      for (int t = 0; t < 8; ++t) {
        float v = st[t][r] * 0.125f + mrow[16 * t];
        st[t][r] = v;
        rm = fmaxf(rm, v);
      }
      mx[r] = rm;
    }
    // ---- cross-lane max within each 16-lane half (rows live per half)
#pragma unroll
    for (int r = 0; r < 8; ++r) {
      mx[r] = fmaxf(mx[r], __shfl_xor(mx[r], 1));
      mx[r] = fmaxf(mx[r], __shfl_xor(mx[r], 2));
      mx[r] = fmaxf(mx[r], __shfl_xor(mx[r], 4));
      mx[r] = fmaxf(mx[r], __shfl_xor(mx[r], 8));
    }
    float cf[8];
#pragma unroll
    for (int r = 0; r < 8; ++r) {
      const float nm = fmaxf(m_run[r], mx[r]);
      cf[r] = __expf(m_run[r] - nm);
      m_run[r] = nm;
    }
    // ---- exponentiate + row sums
    float rs[8];
#pragma unroll
    for (int r = 0; r < 8; ++r) {
      float s = 0.f;
#pragma unroll
      for (int t = 0; t < 8; ++t) {
        const float p = __expf(st[t][r] - m_run[r]);
        st[t][r] = p;
        s += p;
      }
      rs[r] = s;
    }
#pragma unroll
    for (int r = 0; r < 8; ++r) {
      rs[r] += __shfl_xor(rs[r], 1);
      rs[r] += __shfl_xor(rs[r], 2);
      rs[r] += __shfl_xor(rs[r], 4);
      rs[r] += __shfl_xor(rs[r], 8);
      l_run[r] = l_run[r] * cf[r] + rs[r];
    }
    // ---- rescale context accumulators
#pragma unroll
    for (int nt = 0; nt < 4; ++nt)
#pragma unroll
      for (int r = 0; r < 8; ++r) acc[nt][r] *= cf[r];
    // ---- stage P (D-layout -> LDS row-major) for the A-fragment transpose
#pragma unroll
    for (int t = 0; t < 8; ++t)
#pragma unroll
      for (int r = 0; r < 8; ++r) {
        const int m = r + (lh << 3);
        pw[m * 128 + 16 * t + n] = (__bf16)st[t][r];
      }
    // ---- ctx += P @ V  (A from LDS, B from staged V^T)
#pragma unroll
    for (int c = 0; c < 4; ++c) {
      const __bf16* pp = pw + (size_t)n * 128 + 32 * c + (lh << 3);
      v8bf c0 = *(const v8bf*)pp;
      v8bf c1 = *(const v8bf*)(pp + 16);
      v16bf pf;
#pragma unroll
      for (int i = 0; i < 8; ++i) { pf[i] = c0[i]; pf[i + 8] = c1[i]; }
#pragma unroll
      for (int nt = 0; nt < 4; ++nt) {
        v16bf bf = load_b_frag(vbL, 128, 16 * nt, 32 * c, lane);
        acc[nt] = wmma_bf16(pf, bf, acc[nt]);
      }
    }
    tdm_wait_sync();
  }
  // ---- epilogue: normalize, weight, merge heads into (B,S,HID)
  const float w = hw[hh];
#pragma unroll
  for (int nt = 0; nt < 4; ++nt)
#pragma unroll
    for (int r = 0; r < 8; ++r) {
      const int q = qBase + r + (lh << 3);
      const int d = 16 * nt + n;
      ctx_out[((size_t)bi * SS + q) * HIDC + hh * HDC + d] = acc[nt][r] * w / l_run[r];
    }
  if (n == 0) {
#pragma unroll
    for (int r = 0; r < 8; ++r) {
      const int q = qBase + r + (lh << 3);
      stats[((size_t)bh * SS + q) * 2] = m_run[r];
      stats[((size_t)bh * SS + q) * 2 + 1] = l_run[r];
    }
  }
}

// ---------------------------------------------------------------------------
// Kernel 4: emit probs for the last CHUNK rows using saved (m,l) stats.
// Grid (4 key-chunks of 512, NH, B), 256 threads; wave owns 16 q-rows.
// ---------------------------------------------------------------------------
__global__ __launch_bounds__(256) void probs_kernel(const __bf16* __restrict__ qb,
                                                    const __bf16* __restrict__ kbuf,
                                                    const float* __restrict__ mask,
                                                    const float* __restrict__ hw,
                                                    const float* __restrict__ stats,
                                                    float* __restrict__ probs_out) {
  const int lane = threadIdx.x & 31, wave = threadIdx.x >> 5;
  const int kc = blockIdx.x, hh = blockIdx.y, bi = blockIdx.z;
  const int bh = bi * NHC + hh;
  const int qBase = (SS - CHUNKC) + wave * 16;
  const int lh = lane >> 4, n = lane & 15;
  const __bf16* qp = qb + (size_t)bh * SS * HDC;
  const __bf16* kp = kbuf + (size_t)bh * SS * HDC;

  v16bf qf[2];
#pragma unroll
  for (int j = 0; j < 2; ++j) qf[j] = load_a_frag(qp, HDC, qBase, 32 * j, lane);

  float mi[8], li[8];
#pragma unroll
  for (int r = 0; r < 8; ++r) {
    const int q = qBase + r + (lh << 3);
    mi[r] = stats[((size_t)bh * SS + q) * 2];
    li[r] = stats[((size_t)bh * SS + q) * 2 + 1];
  }
  const float w = hw[hh];
  const float* mask_b = mask + (size_t)bi * SS * SS;

  for (int k0 = kc * 512; k0 < kc * 512 + 512; k0 += 16) {
    v8f s = {};
#pragma unroll
    for (int j = 0; j < 2; ++j) {
      v16bf bf = load_b_frag(kp, HDC, k0, 32 * j, lane);
      s = wmma_bf16(qf[j], bf, s);
    }
#pragma unroll
    for (int r = 0; r < 8; ++r) {
      const int q = qBase + r + (lh << 3);
      const float sm = s[r] * 0.125f + mask_b[(size_t)q * SS + k0 + n];
      const float p = __expf(sm - mi[r]) / li[r] * w;
      probs_out[((size_t)bh * CHUNKC + (q - (SS - CHUNKC))) * SS + k0 + n] = p;
    }
  }
}

// ---------------------------------------------------------------------------
// Launch
// ---------------------------------------------------------------------------
extern "C" void kernel_launch(void* const* d_in, const int* in_sizes, int n_in,
                              void* d_out, int out_size, void* d_ws, size_t ws_size,
                              hipStream_t stream) {
  (void)in_sizes; (void)n_in; (void)out_size; (void)ws_size;
  const float* hs   = (const float*)d_in[0];
  const float* mask = (const float*)d_in[1];
  const float* Wq   = (const float*)d_in[2];
  const float* bq   = (const float*)d_in[3];
  const float* Wk   = (const float*)d_in[4];
  const float* bk   = (const float*)d_in[5];
  const float* Wv   = (const float*)d_in[6];
  const float* bv   = (const float*)d_in[7];
  const float* hwv  = (const float*)d_in[8];
  float* out = (float*)d_out;

  char* ws = (char*)d_ws;
  size_t off = 0;
  __bf16* hsb = (__bf16*)(ws + off); off += (size_t)BB * SS * HIDC * sizeof(__bf16);
  __bf16* wqb = (__bf16*)(ws + off); off += (size_t)HIDC * HIDC * sizeof(__bf16);
  __bf16* wkb = (__bf16*)(ws + off); off += (size_t)HIDC * HIDC * sizeof(__bf16);
  __bf16* wvb = (__bf16*)(ws + off); off += (size_t)HIDC * HIDC * sizeof(__bf16);
  __bf16* qbf = (__bf16*)(ws + off); off += (size_t)BB * NHC * SS * HDC * sizeof(__bf16);
  __bf16* kbf = (__bf16*)(ws + off); off += (size_t)BB * NHC * SS * HDC * sizeof(__bf16);
  __bf16* vtb = (__bf16*)(ws + off); off += (size_t)BB * NHC * HDC * SS * sizeof(__bf16);
  float*  stats = (float*)(ws + off); off += (size_t)BB * NHC * SS * 2 * sizeof(float);

  // 1) convert inputs to bf16
  {
    const int nhs = BB * SS * HIDC;
    cvt_bf16_kernel<<<(nhs + 255) / 256, 256, 0, stream>>>(hs, hsb, nhs);
    const int nw = HIDC * HIDC;
    cvt_bf16_kernel<<<(nw + 255) / 256, 256, 0, stream>>>(Wq, wqb, nw);
    cvt_bf16_kernel<<<(nw + 255) / 256, 256, 0, stream>>>(Wk, wkb, nw);
    cvt_bf16_kernel<<<(nw + 255) / 256, 256, 0, stream>>>(Wv, wvb, nw);
  }

  // 2) QKV projections (WMMA GEMM, TDM-staged weight panels)
  {
    dim3 grid(BB * SS / 128, HIDC / 64);
    qkv_gemm_kernel<<<grid, 256, 0, stream>>>(hsb, wqb, bq, qbf, vtb, 0);
    qkv_gemm_kernel<<<grid, 256, 0, stream>>>(hsb, wkb, bk, kbf, vtb, 1);
    qkv_gemm_kernel<<<grid, 256, 0, stream>>>(hsb, wvb, bv, qbf, vtb, 2);
  }

  // 3) fused flash attention -> context_layer (first part of d_out) + stats
  {
    dim3 grid(SS / 128, NHC, BB);
    flash_attn_kernel<<<grid, 256, 0, stream>>>(qbf, kbf, vtb, mask, hwv, out, stats);
  }

  // 4) probs for the last CHUNK rows (second part of d_out)
  {
    dim3 grid(4, NHC, BB);
    probs_kernel<<<grid, 256, 0, stream>>>(qbf, kbf, mask, hwv, stats,
                                           out + (size_t)BB * SS * HIDC);
  }
}
